// QLSTM_65481071409286
// MI455X (gfx1250) — matrix operand order repaired and checked by reading, not compiled, not run on hardware
//
#include <hip/hip_runtime.h>

#define DEVINL __device__ __forceinline__

typedef __attribute__((ext_vector_type(16))) _Float16 v16h;
typedef __attribute__((ext_vector_type(8)))  float    v8f;
typedef __attribute__((ext_vector_type(4)))  unsigned v4u;
typedef __attribute__((ext_vector_type(8)))  unsigned v8u;

constexpr int T = 512, B = 64, E = 256, H = 256;
constexpr int EH = E + H;              // 512
constexpr int TB = T * B;              // 32768 rows

// ---------------------------------------------------------------------------
// WMMA wrapper: D = A(16x32 f16) * B(32x16 f16) + C(16x16 f32)
// ---------------------------------------------------------------------------
DEVINL v8f wmma32(v16h a, v16h b, v8f c) {
  return __builtin_amdgcn_wmma_f32_16x16x32_f16(false, a, false, b, (short)0, c,
                                                false, false);
}

// ---------------------------------------------------------------------------
// Fragment loaders (wave32; A/B share a contiguous k order, C/D as in CDNA5 ISA §7.12.2)
// A 16x32 f16: lane l -> row = base + (l&15); element e=g*8+j -> K = hi*16 + g*8 + j
// B 32x16 f16: lane l -> col = base + (l&15); element e -> K = e + hi*16
// C/D 16x16 f32: lane l, vgpr v -> row = v + hi*8, col = l&15
// ---------------------------------------------------------------------------
DEVINL v16h load_a_f16(const _Float16* M, size_t rstride, int row_base, int k0, int lane) {
  int row = row_base + (lane & 15);
  int hi  = (lane >> 4) & 1;
  const _Float16* p = M + (size_t)row * rstride + k0 + hi * 16;
  v16h a;
#pragma unroll
  for (int g = 0; g < 2; ++g)
#pragma unroll
    for (int j = 0; j < 8; ++j)
      a[g * 8 + j] = p[g * 8 + j];
  return a;
}

DEVINL v16h load_a_f32cvt(const float* M, size_t rstride, int row_base, int k0, int lane) {
  int row = row_base + (lane & 15);
  int hi  = (lane >> 4) & 1;
  const float* p = M + (size_t)row * rstride + k0 + hi * 16;
  v16h a;
#pragma unroll
  for (int g = 0; g < 2; ++g)
#pragma unroll
    for (int j = 0; j < 8; ++j)
      a[g * 8 + j] = (_Float16)p[g * 8 + j];
  return a;
}

DEVINL v16h load_b_f32cvt(const float* M, size_t kstride, int k0, int col_base, int lane) {
  int col = col_base + (lane & 15);
  int hi  = (lane >> 4) & 1;
  const float* p = M + (size_t)(k0 + hi * 16) * kstride + col;
  v16h b;
#pragma unroll
  for (int e = 0; e < 16; ++e) b[e] = (_Float16)p[e * kstride];
  return b;
}

// B fragment from "transposed" storage: 16 contiguous halves per lane
DEVINL v16h load_b_contig(const _Float16* p) {
  v16h b;
#pragma unroll
  for (int e = 0; e < 16; ++e) b[e] = p[e];
  return b;
}

DEVINL float sigmoidf_(float x) { return 1.0f / (1.0f + __expf(-x)); }

DEVINL unsigned lds_off(const void* p) { return (unsigned)(uintptr_t)p; }

// ---------------------------------------------------------------------------
// Kernel 1: prep — f16 conversion of inputs; transpose LSTM weights to
// WT[gate][n][k] (so B-fragments are contiguous 32B runs); zero h state.
// ---------------------------------------------------------------------------
__global__ void k_prep(const float* __restrict__ inputs,
                       const float* __restrict__ Wf, const float* __restrict__ Wi,
                       const float* __restrict__ Wu, const float* __restrict__ Wo,
                       _Float16* __restrict__ inph, _Float16* __restrict__ WT,
                       float* __restrict__ hxA, float* __restrict__ hxB,
                       unsigned* __restrict__ bar) {
  size_t stride = (size_t)gridDim.x * blockDim.x;
  size_t i0 = (size_t)blockIdx.x * blockDim.x + threadIdx.x;
  for (size_t i = i0; i < (size_t)T * B * E; i += stride) inph[i] = (_Float16)inputs[i];
  const size_t WN = (size_t)EH * H;
  for (size_t i = i0; i < WN; i += stride) {
    size_t k = i / H, n = i % H;
    size_t o = n * EH + k;                  // transposed [n][k]
    WT[0 * WN + o] = (_Float16)Wf[i];
    WT[1 * WN + o] = (_Float16)Wi[i];
    WT[2 * WN + o] = (_Float16)Wu[i];
    WT[3 * WN + o] = (_Float16)Wo[i];
  }
  for (size_t i = i0; i < (size_t)B * H; i += stride) { hxA[i] = 0.f; hxB[i] = 0.f; }
  if (i0 == 0) { bar[0] = 0u; bar[1] = 0u; }
}

// ---------------------------------------------------------------------------
// Kernel 2: q/k projections. M=T*B, K=E, N=E, one 16x16 tile per wave.
// ---------------------------------------------------------------------------
__global__ void k_qk(const float* __restrict__ inputs,
                     const float* __restrict__ rot, const float* __restrict__ ent,
                     _Float16* __restrict__ qh, _Float16* __restrict__ kh) {
  int wave = blockIdx.x * (blockDim.x >> 5) + (threadIdx.x >> 5);
  int lane = threadIdx.x & 31;
  int mt = wave & (TB / 16 - 1);
  int r  = wave >> 11;
  int nt = r & 15;
  int which = r >> 4;
  const float* W = which ? ent : rot;
  _Float16* Out  = which ? kh : qh;

  v8f acc = {};
  for (int k0 = 0; k0 < E; k0 += 32) {
    v16h a = load_a_f32cvt(inputs, E, mt * 16, k0, lane);
    v16h b = load_b_f32cvt(W, E, k0, nt * 16, lane);
    acc = wmma32(a, b, acc);
  }
  int nn = lane & 15, hi = (lane >> 4) & 1;
#pragma unroll
  for (int v = 0; v < 8; ++v) {
    int m = mt * 16 + v + hi * 8;
    Out[(size_t)m * E + nt * 16 + nn] = (_Float16)acc[v];
  }
}

// ---------------------------------------------------------------------------
// Kernel 3: flash attention. Block = 4 waves sharing ONE batch b; each wave
// owns a 16-row t-tile with Q resident in registers. K/V s-chunks staged into
// double-buffered LDS via global_load_async_to_lds_b128; the async DMA for
// chunk i+1 runs underneath chunk i's WMMAs (ASYNCcnt ping-pong).
// ---------------------------------------------------------------------------
__global__ void k_attn(const _Float16* __restrict__ qh, const _Float16* __restrict__ kh,
                       const _Float16* __restrict__ inph, _Float16* __restrict__ ctxh) {
  __shared__ __align__(16) _Float16 kch[2][32][256];   // 2 x 16KB [s_local][f]
  __shared__ __align__(16) _Float16 vch[2][32][256];   // 2 x 16KB [s_local][e]
  __shared__ __align__(16) _Float16 plds[4][16 * 32];  // per-wave P transpose slice

  int w    = threadIdx.x >> 5;
  int lane = threadIdx.x & 31;
  int b  = blockIdx.x >> 3;                 // 64 batches
  int tt = (blockIdx.x & 7) * 4 + w;        // 32 t-tiles
  int nn = lane & 15, hi = (lane >> 4) & 1;

  const _Float16* qb = qh + (size_t)b * E;  // rows = t, stride B*E
  const float scale = 0.0625f;              // 1/sqrt(256)

  // issue one 32-row K/V stage (1024 b128 chunks each) into buffer `buf`
  auto stage = [&](int buf, int s0) {
    for (int i = threadIdx.x; i < 1024; i += 128) {
      int row = i >> 5;                     // s_local
      int c16 = i & 31;                     // 16B chunk in row
      unsigned dk = lds_off(&kch[buf][row][0]) + c16 * 16;
      unsigned dv = lds_off(&vch[buf][row][0]) + c16 * 16;
      const char* sk = (const char*)(kh   + ((size_t)(s0 + row) * B + b) * E) + c16 * 16;
      const char* sv = (const char*)(inph + ((size_t)(s0 + row) * B + b) * E) + c16 * 16;
      asm volatile("global_load_async_to_lds_b128 %0, %1, off"
                   :: "v"(dk), "v"(sk) : "memory");
      asm volatile("global_load_async_to_lds_b128 %0, %1, off"
                   :: "v"(dv), "v"(sv) : "memory");
    }
  };

  // ---- Q tile resident in registers: 8 A-fragments (16x256 f16) ----
  v16h qfr[8];
#pragma unroll
  for (int f = 0; f < 8; ++f)
    qfr[f] = load_a_f16(qb, (size_t)B * E, tt * 16, f * 32, lane);

  v8f ctx[16];
#pragma unroll
  for (int ntile = 0; ntile < 16; ++ntile) ctx[ntile] = (v8f){};
  float mrow[8], lrow[8];
#pragma unroll
  for (int v = 0; v < 8; ++v) { mrow[v] = -1e30f; lrow[v] = 0.f; }

  stage(0, 0);                              // prologue: fill buffer 0
  for (int s0 = 0; s0 < T; s0 += 32) {
    int buf = (s0 >> 5) & 1;
    asm volatile("s_wait_asynccnt 0x0" ::: "memory");
    __syncthreads();                        // staged data visible to all waves
    if (s0 + 32 < T) stage(buf ^ 1, s0 + 32);   // overlap next DMA with compute

    // ---- scores: two 16x16 tiles over contraction f ----
    v8f S0 = {}, S1 = {};
#pragma unroll
    for (int f = 0; f < 8; ++f) {
      v16h kb0 = load_b_contig(&kch[buf][nn][f * 32 + hi * 16]);
      v16h kb1 = load_b_contig(&kch[buf][nn + 16][f * 32 + hi * 16]);
      S0 = wmma32(qfr[f], kb0, S0);
      S1 = wmma32(qfr[f], kb1, S1);
    }
    // ---- online softmax ----
#pragma unroll
    for (int v = 0; v < 8; ++v) {
      float s0v = S0[v] * scale, s1v = S1[v] * scale;
      float mx = fmaxf(s0v, s1v);
#pragma unroll
      for (int off = 1; off < 16; off <<= 1) mx = fmaxf(mx, __shfl_xor(mx, off, 32));
      float mnew = fmaxf(mrow[v], mx);
      float sc = __expf(mrow[v] - mnew);
      float p0 = __expf(s0v - mnew);
      float p1 = __expf(s1v - mnew);
      float rs = p0 + p1;
#pragma unroll
      for (int off = 1; off < 16; off <<= 1) rs += __shfl_xor(rs, off, 32);
      lrow[v] = lrow[v] * sc + rs;
      mrow[v] = mnew;
#pragma unroll
      for (int ntile = 0; ntile < 16; ++ntile) ctx[ntile][v] *= sc;
      int m = v + hi * 8;
      plds[w][m * 32 + nn]      = (_Float16)p0;
      plds[w][m * 32 + nn + 16] = (_Float16)p1;
    }
    // P slice is wave-private: DS ops are in-order within a wave; only a
    // compiler scheduling fence is needed (no block-wide barrier).
    __builtin_amdgcn_wave_barrier();
    // ---- P: C-layout -> A-fragment via LDS ----
    v16h pa;
#pragma unroll
    for (int g = 0; g < 2; ++g)
#pragma unroll
      for (int j = 0; j < 8; ++j)
        pa[g * 8 + j] = plds[w][(lane & 15) * 32 + hi * 16 + g * 8 + j];
    // ---- ctx += P * V (V from LDS, B[k=s_local][n=e]) ----
#pragma unroll
    for (int ntile = 0; ntile < 16; ++ntile) {
      v16h vf;
#pragma unroll
      for (int e = 0; e < 16; ++e) vf[e] = vch[buf][hi * 16 + e][ntile * 16 + nn];
      ctx[ntile] = wmma32(pa, vf, ctx[ntile]);
    }
  }
  // ---- normalize and store ctxh[t,b,e] ----
#pragma unroll
  for (int ntile = 0; ntile < 16; ++ntile)
#pragma unroll
    for (int v = 0; v < 8; ++v) {
      int t = tt * 16 + v + hi * 8;
      float val = ctx[ntile][v] / lrow[v];
      ctxh[((size_t)t * B + b) * E + ntile * 16 + nn] = (_Float16)val;
    }
}

// ---------------------------------------------------------------------------
// Kernel 4: ctx_proj = ctx @ Wc + bc   (M=T*B, K=E, N=H) -> cp f32
// ---------------------------------------------------------------------------
__global__ void k_proj(const _Float16* __restrict__ ctxh,
                       const float* __restrict__ Wc, const float* __restrict__ bc,
                       float* __restrict__ cp) {
  int wave = blockIdx.x * (blockDim.x >> 5) + (threadIdx.x >> 5);
  int lane = threadIdx.x & 31;
  int mt = wave & (TB / 16 - 1);
  int nt = wave >> 11;
  v8f acc = {};
  for (int k0 = 0; k0 < E; k0 += 32) {
    v16h a = load_a_f16(ctxh, E, mt * 16, k0, lane);
    v16h b = load_b_f32cvt(Wc, H, k0, nt * 16, lane);
    acc = wmma32(a, b, acc);
  }
  int nn = lane & 15, hi = (lane >> 4) & 1;
  float bias = bc[nt * 16 + nn];
#pragma unroll
  for (int v = 0; v < 8; ++v) {
    int m = mt * 16 + v + hi * 8;
    cp[(size_t)m * H + nt * 16 + nn] = acc[v] + bias;
  }
}

// ---------------------------------------------------------------------------
// Global sense-reversing barrier (16 blocks, persistent LSTM kernel)
// ---------------------------------------------------------------------------
DEVINL void grid_barrier(unsigned* bar) {
  __syncthreads();
  if (threadIdx.x == 0) {
    __threadfence();
    unsigned gen = __hip_atomic_load(&bar[1], __ATOMIC_ACQUIRE, __HIP_MEMORY_SCOPE_AGENT);
    unsigned arrived =
        __hip_atomic_fetch_add(&bar[0], 1u, __ATOMIC_ACQ_REL, __HIP_MEMORY_SCOPE_AGENT);
    if (arrived == gridDim.x - 1) {
      __hip_atomic_store(&bar[0], 0u, __ATOMIC_RELAXED, __HIP_MEMORY_SCOPE_AGENT);
      __hip_atomic_fetch_add(&bar[1], 1u, __ATOMIC_ACQ_REL, __HIP_MEMORY_SCOPE_AGENT);
    } else {
      while (__hip_atomic_load(&bar[1], __ATOMIC_ACQUIRE, __HIP_MEMORY_SCOPE_AGENT) == gen)
        __builtin_amdgcn_s_sleep(1);
    }
    __threadfence();
  }
  __syncthreads();
}

// combined = [ x(t) | hprev + cp(t) ] as A-fragment (rows = batch, K over E+H)
DEVINL v16h load_comb(const float* __restrict__ inputs, const float* __restrict__ hprev,
                      const float* __restrict__ cp, int t, int row_base, int k0, int lane) {
  int bb = row_base + (lane & 15);
  int hi = (lane >> 4) & 1;
  v16h a;
#pragma unroll
  for (int g = 0; g < 2; ++g)
#pragma unroll
    for (int j = 0; j < 8; ++j) {
      int k = k0 + hi * 16 + g * 8 + j;
      float val;
      if (k < E) {
        val = inputs[(size_t)t * B * E + (size_t)bb * E + k];
      } else {
        int kk = k - E;
        val = hprev[(size_t)bb * H + kk] + cp[(size_t)t * B * H + (size_t)bb * H + kk];
      }
      a[g * 8 + j] = (_Float16)val;
    }
  return a;
}

// ---------------------------------------------------------------------------
// Kernel 5: persistent LSTM. 16 blocks (one h-tile nt each) x 4 waves (one
// batch tile mt each). Block stages its 64KB 4-gate weight panel into LDS
// ONCE via the Tensor Data Mover, then runs 512 steps reading weights from
// LDS. c state in registers; h double-buffered; one grid barrier per step.
// ---------------------------------------------------------------------------
__global__ void k_lstm(const float* __restrict__ inputs, const float* __restrict__ cp,
                       const _Float16* __restrict__ WT,
                       const float* __restrict__ bf, const float* __restrict__ bi,
                       const float* __restrict__ bu, const float* __restrict__ bo,
                       float* __restrict__ hxA, float* __restrict__ hxB,
                       float* __restrict__ out, float* __restrict__ hx_out,
                       float* __restrict__ cx_out, unsigned* __restrict__ bar) {
  __shared__ __align__(16) _Float16 wlds[4][16][EH];   // [gate][col][k] = 64KB

  int lane = threadIdx.x & 31;
  int mt = threadIdx.x >> 5;              // 4 batch tiles (wave id)
  int nt = blockIdx.x;                    // 16 h tiles
  int nn = lane & 15, hi = (lane >> 4) & 1;
  int hh = nt * 16 + nn;

  // ---- TDM: stage 4 x 16KB contiguous weight slices (WT[gate][nt*16..+16][*]) ----
  if (threadIdx.x < 32) {
    const size_t WN = (size_t)EH * H;
#pragma unroll
    for (int gate = 0; gate < 4; ++gate) {
      unsigned long long gaddr =
          (unsigned long long)(uintptr_t)(WT + gate * WN + (size_t)nt * 16 * EH);
      unsigned laddr = lds_off(&wlds[gate][0][0]);
      v4u g0;
      g0[0] = 1u;                                   // count=1, user mode
      g0[1] = laddr;                                // lds_addr
      g0[2] = (unsigned)(gaddr & 0xFFFFFFFFu);      // global_addr[31:0]
      g0[3] = (unsigned)((gaddr >> 32) & 0x1FFFFFFu) | (2u << 30);  // addr[56:32] | type=2
      v8u g1;
      g1[0] = 1u << 16;                             // data_size=1 (2 bytes)
      g1[1] = (8192u & 0xFFFFu) << 16;              // tensor_dim0[15:0] @ bits 63:48
      g1[2] = (8192u >> 16) | (1u << 16);           // tensor_dim0 hi | tensor_dim1=1
      g1[3] = 8192u << 16;                          // tile_dim0 @ bits 127:112
      g1[4] = 1u;                                   // tile_dim1 = 1
      g1[5] = 8192u;                                // tensor_dim0_stride low
      g1[6] = 0u;
      g1[7] = 0u;
      asm volatile("tensor_load_to_lds %0, %1" :: "s"(g0), "s"(g1) : "memory");
    }
    __builtin_amdgcn_s_wait_tensorcnt(0);
  }
  __syncthreads();

  float creg[8];
#pragma unroll
  for (int v = 0; v < 8; ++v) creg[v] = 0.f;
  float bfv = bf[hh], biv = bi[hh], buv = bu[hh], bov = bo[hh];

  for (int t = 0; t < T; ++t) {
    const float* hprev = (t & 1) ? hxB : hxA;
    float*       hnext = (t & 1) ? hxA : hxB;

    if (t + 1 < T) {  // gfx1250 global_prefetch of next step's activation slabs
      __builtin_prefetch(inputs + (size_t)(t + 1) * B * E + (size_t)mt * 16 * E, 0, 1);
      __builtin_prefetch(cp + (size_t)(t + 1) * B * H + (size_t)mt * 16 * H, 0, 1);
    }

    v8f aF = {}, aI = {}, aG = {}, aO = {};
    for (int k0 = 0; k0 < EH; k0 += 32) {
      v16h a = load_comb(inputs, hprev, cp, t, mt * 16, k0, lane);
      aF = wmma32(a, load_b_contig(&wlds[0][nn][k0 + hi * 16]), aF);
      aI = wmma32(a, load_b_contig(&wlds[1][nn][k0 + hi * 16]), aI);
      aG = wmma32(a, load_b_contig(&wlds[2][nn][k0 + hi * 16]), aG);
      aO = wmma32(a, load_b_contig(&wlds[3][nn][k0 + hi * 16]), aO);
    }
#pragma unroll
    for (int v = 0; v < 8; ++v) {
      int bb = mt * 16 + v + hi * 8;
      float f = sigmoidf_(aF[v] + bfv);
      float i = sigmoidf_(aI[v] + biv);
      float g = tanhf(aG[v] + buv);
      float o = sigmoidf_(aO[v] + bov);
      float c = f * creg[v] + i * g;
      float h = o * tanhf(c);
      creg[v] = c;
      hnext[(size_t)bb * H + hh] = h;
      out[(size_t)t * B * H + (size_t)bb * H + hh] = h;
      if (t == T - 1) {
        hx_out[(size_t)bb * H + hh] = h;
        cx_out[(size_t)bb * H + hh] = c;
      }
    }
    grid_barrier(bar);
  }
}

// ---------------------------------------------------------------------------
// Host launcher
// ---------------------------------------------------------------------------
extern "C" void kernel_launch(void* const* d_in, const int* in_sizes, int n_in,
                              void* d_out, int out_size, void* d_ws, size_t ws_size,
                              hipStream_t stream) {
  const float* inputs = (const float*)d_in[0];
  const float* rot    = (const float*)d_in[1];
  const float* ent    = (const float*)d_in[2];
  const float* Wf = (const float*)d_in[3];  const float* bf = (const float*)d_in[4];
  const float* Wi = (const float*)d_in[5];  const float* bi = (const float*)d_in[6];
  const float* Wu = (const float*)d_in[7];  const float* bu = (const float*)d_in[8];
  const float* Wo = (const float*)d_in[9];  const float* bo = (const float*)d_in[10];
  const float* Wc = (const float*)d_in[11]; const float* bc = (const float*)d_in[12];

  float* out    = (float*)d_out;
  float* hx_out = out + (size_t)T * B * H;
  float* cx_out = hx_out + (size_t)B * H;

  char* w = (char*)d_ws;
  size_t off = 0;
  auto take = [&](size_t bytes) { size_t o = off; off += (bytes + 255) & ~(size_t)255; return o; };
  const size_t halfTBE = (size_t)T * B * E * sizeof(_Float16);
  _Float16* inph = (_Float16*)(w + take(halfTBE));
  _Float16* qh   = (_Float16*)(w + take(halfTBE));
  _Float16* kh   = (_Float16*)(w + take(halfTBE));
  _Float16* ctxh = (_Float16*)(w + take(halfTBE));
  _Float16* WT   = (_Float16*)(w + take((size_t)4 * EH * H * sizeof(_Float16)));
  float*    cp   = (float*)(w + take((size_t)T * B * H * sizeof(float)));
  float*    hxA  = (float*)(w + take((size_t)B * H * sizeof(float)));
  float*    hxB  = (float*)(w + take((size_t)B * H * sizeof(float)));
  unsigned* bar  = (unsigned*)(w + take(256));

  k_prep<<<2048, 256, 0, stream>>>(inputs, Wf, Wi, Wu, Wo, inph, WT, hxA, hxB, bar);
  k_qk<<<8192, 256, 0, stream>>>(inputs, rot, ent, qh, kh);
  k_attn<<<512, 128, 0, stream>>>(qh, kh, inph, ctxh);
  k_proj<<<4096, 256, 0, stream>>>(ctxh, Wc, bc, cp);
  k_lstm<<<16, 128, 0, stream>>>(inputs, cp, WT, bf, bi, bu, bo,
                                 hxA, hxB, out, hx_out, cx_out, bar);
  (void)in_sizes; (void)n_in; (void)out_size; (void)ws_size;
}
